// BitLinear_89687507076318
// MI455X (gfx1250) — 8-bit matrix operand order repaired and checked by reading, compiled, not run on hardware
//
#include <hip/hip_runtime.h>

typedef __attribute__((ext_vector_type(8))) int v8i;

#define K_DIM 2048
#define N_DIM 2048
#define EPSQ 1e-8f

#define OFF_PART  256
#define OFF_SCALE 32768
#define OFF_WQ    (1u << 20)
#define OFF_XQ    (8u << 20)

__global__ __launch_bounds__(256) void reduceAbsW(const float* __restrict__ w,
                                                  float* __restrict__ part) {
  __shared__ float red[256];
  const int idx = blockIdx.x * 256 + threadIdx.x;
  float4 v = ((const float4*)w)[idx];
  float s = fabsf(v.x) + fabsf(v.y) + fabsf(v.z) + fabsf(v.w);
  red[threadIdx.x] = s;
  __syncthreads();
  for (int off = 128; off > 0; off >>= 1) {
    if (threadIdx.x < off) red[threadIdx.x] += red[threadIdx.x + off];
    __syncthreads();
  }
  if (threadIdx.x == 0) part[blockIdx.x] = red[0];
}

__global__ __launch_bounds__(256) void finalizeSum(const float* __restrict__ part,
                                                   float* __restrict__ sum) {
  __shared__ float red[256];
  float s = 0.f;
  #pragma unroll
  for (int i = 0; i < 16; ++i) s += part[threadIdx.x + i * 256];
  red[threadIdx.x] = s;
  __syncthreads();
  for (int off = 128; off > 0; off >>= 1) {
    if (threadIdx.x < off) red[threadIdx.x] += red[threadIdx.x + off];
    __syncthreads();
  }
  if (threadIdx.x == 0) sum[0] = red[0];
}

__device__ __forceinline__ int quantClamp(float v, float inv, float lim) {
  float r = rintf(v * inv);
  r = fminf(fmaxf(r, -lim), lim);
  return ((int)r) & 0xFF;
}

__global__ __launch_bounds__(256) void quantW(const float* __restrict__ w,
                                              const float* __restrict__ sumAbs,
                                              char* __restrict__ wq) {
  const float alpha = fmaxf(sumAbs[0] * (1.0f / 4194304.0f), EPSQ);
  const float inv = 1.0f / alpha;
  const int idx = blockIdx.x * 256 + threadIdx.x;
  float4 v = ((const float4*)w)[idx];
  int p = quantClamp(v.x, inv, 1.f) | (quantClamp(v.y, inv, 1.f) << 8) |
          (quantClamp(v.z, inv, 1.f) << 16) | (quantClamp(v.w, inv, 1.f) << 24);
  ((int*)wq)[idx] = p;
}

__global__ __launch_bounds__(256) void quantX(const float* __restrict__ x,
                                              const float* __restrict__ sumAbs,
                                              char* __restrict__ xq,
                                              float* __restrict__ scale) {
  __shared__ float red[256];
  const int token = blockIdx.x;
  const float* row = x + (size_t)token * K_DIM;
  float4 v0 = ((const float4*)row)[threadIdx.x * 2 + 0];
  float4 v1 = ((const float4*)row)[threadIdx.x * 2 + 1];
  float m = fmaxf(fmaxf(fmaxf(fabsf(v0.x), fabsf(v0.y)), fmaxf(fabsf(v0.z), fabsf(v0.w))),
                  fmaxf(fmaxf(fabsf(v1.x), fabsf(v1.y)), fmaxf(fabsf(v1.z), fabsf(v1.w))));
  red[threadIdx.x] = m;
  __syncthreads();
  for (int off = 128; off > 0; off >>= 1) {
    if (threadIdx.x < off) red[threadIdx.x] = fmaxf(red[threadIdx.x], red[threadIdx.x + off]);
    __syncthreads();
  }
  const float beta = fmaxf(red[0] * (1.0f / 127.0f), EPSQ);
  if (threadIdx.x == 0) {
    const float alpha = fmaxf(sumAbs[0] * (1.0f / 4194304.0f), EPSQ);
    scale[token] = alpha * beta;
  }
  const float inv = 1.0f / beta;
  int lo = quantClamp(v0.x, inv, 127.f) | (quantClamp(v0.y, inv, 127.f) << 8) |
           (quantClamp(v0.z, inv, 127.f) << 16) | (quantClamp(v0.w, inv, 127.f) << 24);
  int hi = quantClamp(v1.x, inv, 127.f) | (quantClamp(v1.y, inv, 127.f) << 8) |
           (quantClamp(v1.z, inv, 127.f) << 16) | (quantClamp(v1.w, inv, 127.f) << 24);
  ((int2*)(xq + (size_t)token * K_DIM))[threadIdx.x] = make_int2(lo, hi);
}

__device__ __forceinline__ v8i loadA(const char* p, int h) {
  v8i a;
  #pragma unroll
  for (int c = 0; c < 4; ++c) {
    int2 av = *(const int2*)(p + (c >> 1) * 32 + (c & 1) * 8 + h * 16);
    a[2 * c]     = av.x;
    a[2 * c + 1] = av.y;
  }
  return a;
}

__device__ __forceinline__ v8i loadB(const char* p) {
  int4 lo = *(const int4*)(p);
  int4 hi = *(const int4*)(p + 32);
  v8i b = {lo.x, lo.y, lo.z, lo.w, hi.x, hi.y, hi.z, hi.w};
  return b;
}

#define WMMA_IU8(A, B, C) \
  __builtin_amdgcn_wmma_i32_16x16x64_iu8(true, (A), true, (B), (C), false, false)

__global__ __launch_bounds__(256) void gemm_iu8(const char* __restrict__ xq,
                                                const char* __restrict__ wq,
                                                const float* __restrict__ scale,
                                                float* __restrict__ y) {
  const int tid  = threadIdx.x;
  const int wave = tid >> 5;
  const int lane = tid & 31;
  const int l16  = lane & 15;
  const int h    = lane >> 4;

  const int mBase = blockIdx.y * 128 + wave * 16;
  const int nBase = blockIdx.x * 64;

  v8i acc0 = {}, acc1 = {}, acc2 = {}, acc3 = {};

  const char* arow = xq + (size_t)(mBase + l16) * K_DIM;
  const char* b0 = wq + (size_t)(nBase +  0 + l16) * K_DIM + h * 16;
  const char* b1 = wq + (size_t)(nBase + 16 + l16) * K_DIM + h * 16;
  const char* b2 = wq + (size_t)(nBase + 32 + l16) * K_DIM + h * 16;
  const char* b3 = wq + (size_t)(nBase + 48 + l16) * K_DIM + h * 16;

  v8i a  = loadA(arow, h);
  v8i p0 = loadB(b0), p1 = loadB(b1), p2 = loadB(b2), p3 = loadB(b3);

  for (int kk = 0; kk < K_DIM; kk += 128) {
    {
      const int kn = kk + 64;
      __builtin_prefetch(arow + kk + 256, 0, 1);
      v8i an  = loadA(arow + kn, h);
      v8i q0 = loadB(b0 + kn), q1 = loadB(b1 + kn);
      v8i q2 = loadB(b2 + kn), q3 = loadB(b3 + kn);

      acc0 = WMMA_IU8(a, p0, acc0);
      acc1 = WMMA_IU8(a, p1, acc1);
      acc2 = WMMA_IU8(a, p2, acc2);
      acc3 = WMMA_IU8(a, p3, acc3);

      a = an; p0 = q0; p1 = q1; p2 = q2; p3 = q3;
    }
    {
      const int kn = (kk + 128 < K_DIM) ? kk + 128 : kk;
      v8i an  = loadA(arow + kn, h);
      v8i q0 = loadB(b0 + kn), q1 = loadB(b1 + kn);
      v8i q2 = loadB(b2 + kn), q3 = loadB(b3 + kn);

      acc0 = WMMA_IU8(a, p0, acc0);
      acc1 = WMMA_IU8(a, p1, acc1);
      acc2 = WMMA_IU8(a, p2, acc2);
      acc3 = WMMA_IU8(a, p3, acc3);

      a = an; p0 = q0; p1 = q1; p2 = q2; p3 = q3;
    }
  }

  float sc[8];
  #pragma unroll
  for (int r = 0; r < 8; ++r) sc[r] = scale[mBase + r + h * 8];

  v8i accs[4] = {acc0, acc1, acc2, acc3};
  #pragma unroll
  for (int nb = 0; nb < 4; ++nb) {
    const int n = nBase + nb * 16 + l16;
    #pragma unroll
    for (int r = 0; r < 8; ++r) {
      const int m = mBase + r + h * 8;
      y[(size_t)m * N_DIM + n] = (float)accs[nb][r] * sc[r];
    }
  }
}

extern "C" void kernel_launch(void* const* d_in, const int* in_sizes, int n_in,
                              void* d_out, int out_size, void* d_ws, size_t ws_size,
                              hipStream_t stream) {
  const float* x = (const float*)d_in[0];
  const float* w = (const float*)d_in[1];
  float* out = (float*)d_out;
  char* ws = (char*)d_ws;

  const int M = in_sizes[0] / K_DIM;

  float* sumAbs = (float*)(ws);
  float* part   = (float*)(ws + OFF_PART);
  float* scale  = (float*)(ws + OFF_SCALE);
  char*  wq     = ws + OFF_WQ;
  char*  xq     = ws + OFF_XQ;

  reduceAbsW<<<4096, 256, 0, stream>>>(w, part);
  finalizeSum<<<1, 256, 0, stream>>>(part, sumAbs);
  quantW<<<4096, 256, 0, stream>>>(w, sumAbs, wq);
  quantX<<<M, 256, 0, stream>>>(x, sumAbs, xq, scale);
  gemm_iu8<<<dim3(N_DIM / 64, M / 128), 256, 0, stream>>>(xq, wq, scale, out);
}
